// LucyRNN_63316407877920
// MI455X (gfx1250) — matrix operand order repaired and checked by reading, not compiled, not run on hardware
//
#include <hip/hip_runtime.h>
#include <stdint.h>

// ---------------- model dims ----------------
#define BSZ   8
#define TLEN  1024
#define INDIM 80
#define DM    1024
#define VOUT  5000
#define MROWS (BSZ * TLEN)   // 8192 tokens

// ---------------- WMMA types ----------------
typedef __attribute__((ext_vector_type(16))) __bf16       v16bf;
typedef __attribute__((ext_vector_type(8)))  float        v8f;
typedef __attribute__((ext_vector_type(4)))  unsigned int v4u;

union FragAB { v4u u[2]; v16bf b; };

__device__ __forceinline__ uint16_t f2bf(float f) {
  union { float f; uint32_t u; } c; c.f = f;
  uint32_t u = c.u;
  uint32_t r = u + 0x7FFFu + ((u >> 16) & 1u);   // round-to-nearest-even
  return (uint16_t)(r >> 16);
}
__device__ __forceinline__ float bf2f(uint16_t h) {
  union { uint32_t u; float f; } c; c.u = ((uint32_t)h) << 16;
  return c.f;
}

// ---- CDNA5 async global->LDS copy (ASYNCcnt-tracked) ----
__device__ __forceinline__ void async_g2l_b128(uint32_t ldsOff, const void* g) {
  asm volatile("global_load_async_to_lds_b128 %0, %1, off"
               :: "v"(ldsOff), "v"(g) : "memory");
}
__device__ __forceinline__ void wait_async_0() {
  asm volatile("s_wait_asynccnt 0x0" ::: "memory");
}
__device__ __forceinline__ void wait_async_5() {
  asm volatile("s_wait_asynccnt 0x5" ::: "memory");
}

// ============================================================
// GEMM: C[M,N] = A[M,Kp](bf16,row-major) @ Wpacked + bias
// Workgroup (8 waves) computes 64x256; wave = 32x64 tile.
// Double-buffered LDS staging via global_load_async_to_lds_b128.
// A frag (16-bit A 16x32, contiguous k order, same as B): lane L (m=L&15,h=L>>4) holds K chunks
//   [16h,16h+8) and [16h+8,16h+16)  -> two 16B LDS reads.
// B packed per K-step: 16 tiles (N=256) contiguous, 16KB; within a
//   tile lane L holds 16 bf16: element e <-> k = 32*kt + 16*(L>>4)+e,
//   n = tile*16 + (L&15).
// ============================================================
__global__ __launch_bounds__(256)
void gemm_bf16_kernel(const uint16_t* __restrict__ A,
                      const uint16_t* __restrict__ Bp,
                      const float* __restrict__ bias,
                      float* __restrict__ C,
                      int Kp, int N, int nG) {
  __shared__ uint16_t sA[2][64 * 32];    // 2 x 4KB
  __shared__ uint16_t sB[2][16 * 512];   // 2 x 16KB

  const int tid  = threadIdx.x;
  const int lane = tid & 31;
  const int w    = tid >> 5;
  const int mtl  = w & 1;        // wave M sub-tile (0..1)
  const int ntl  = w >> 1;       // wave N sub-tile (0..3)
  const int wgm  = blockIdx.x / nG;
  const int wgn  = blockIdx.x % nG;
  const int m0   = wgm * 64;
  const int KT   = Kp >> 5;
  const int h    = lane >> 4;
  const int l16  = lane & 15;

  // per-thread async-copy assignments
  const int arow   = tid >> 2;   // 0..63
  const int achunk = tid & 3;    // 0..3  (16B chunks of a 64B row)
  const uint16_t* aSrc = A + (size_t)(m0 + arow) * Kp + achunk * 8;
  const uint16_t* bSrc = Bp + (size_t)wgn * KT * 8192 + tid * 8;

  v8f acc[8];
#pragma unroll
  for (int i = 0; i < 8; ++i) acc[i] = (v8f){0.f,0.f,0.f,0.f,0.f,0.f,0.f,0.f};

  // stage 0
  {
    uint32_t aoff = (uint32_t)(uintptr_t)&sA[0][arow * 32 + achunk * 8];
    async_g2l_b128(aoff, aSrc);
#pragma unroll
    for (int i = 0; i < 4; ++i) {
      uint32_t boff = (uint32_t)(uintptr_t)&sB[0][(i * 256 + tid) * 8];
      async_g2l_b128(boff, bSrc + i * 2048);
    }
  }

  for (int kt = 0; kt < KT; ++kt) {
    const int buf = kt & 1;
    if (kt + 1 < KT) {
      // prefetch next K-step into the other buffer
      uint32_t aoff = (uint32_t)(uintptr_t)&sA[buf ^ 1][arow * 32 + achunk * 8];
      async_g2l_b128(aoff, aSrc + (kt + 1) * 32);
      const uint16_t* bs = bSrc + (size_t)(kt + 1) * 8192;
#pragma unroll
      for (int i = 0; i < 4; ++i) {
        uint32_t boff = (uint32_t)(uintptr_t)&sB[buf ^ 1][(i * 256 + tid) * 8];
        async_g2l_b128(boff, bs + i * 2048);
      }
      wait_async_5();            // stage kt's 5 loads complete (in-order)
    } else {
      wait_async_0();
    }
    __syncthreads();             // all waves' stage-kt data resident in LDS

    FragAB a0, a1;
    const uint16_t* sa = &sA[buf][0];
    a0.u[0] = *(const v4u*)(sa + (mtl * 32 + l16) * 32 + 16 * h);
    a0.u[1] = *(const v4u*)(sa + (mtl * 32 + l16) * 32 + 16 * h + 8);
    a1.u[0] = *(const v4u*)(sa + (mtl * 32 + l16 + 16) * 32 + 16 * h);
    a1.u[1] = *(const v4u*)(sa + (mtl * 32 + l16 + 16) * 32 + 16 * h + 8);
#pragma unroll
    for (int jn = 0; jn < 4; ++jn) {
      const uint16_t* sb = &sB[buf][(ntl * 4 + jn) * 512 + lane * 16];
      FragAB bf;
      bf.u[0] = *(const v4u*)(sb);
      bf.u[1] = *(const v4u*)(sb + 8);
      acc[jn]     = __builtin_amdgcn_wmma_f32_16x16x32_bf16(
                        false, a0.b, false, bf.b, (short)0, acc[jn], false, false);
      acc[4 + jn] = __builtin_amdgcn_wmma_f32_16x16x32_bf16(
                        false, a1.b, false, bf.b, (short)0, acc[4 + jn], false, false);
    }
    __syncthreads();             // done reading buf before it is refilled
  }

  // C/D layout: VGPR r -> row base + r + 8*h ; col lane&15
  const int mw = m0 + mtl * 32;
#pragma unroll
  for (int jn = 0; jn < 4; ++jn) {
    int col = wgn * 256 + ntl * 64 + jn * 16 + l16;
    if (col < N) {
      float bv = bias ? bias[col] : 0.f;
#pragma unroll
      for (int r = 0; r < 8; ++r) {
        int row0 = mw + 8 * h + r;
        C[(size_t)row0 * N + col] = acc[jn][r] + bv;
        int row1 = mw + 16 + 8 * h + r;
        C[(size_t)row1 * N + col] = acc[4 + jn][r] + bv;
      }
    }
  }
}

// ---- pack W[K,N] f32 -> bf16 fragment-ordered tiles (zero padded) ----
// flat = ((g*KT + kt)*16 + j)*512 + lane*16 + e ; g = N-group of 256
__global__ void pack_w_kernel(const float* __restrict__ W, uint16_t* __restrict__ Bp,
                              int K, int N, int Kp, int Np) {
  int total = Kp * Np;
  int i = blockIdx.x * 256 + threadIdx.x;
  if (i >= total) return;
  int KT = Kp >> 5;
  int e = i & 15;
  int lane = (i >> 4) & 31;
  int tj = i >> 9;
  int j  = tj & 15;
  int kt = (tj >> 4) % KT;
  int g  = (tj >> 4) / KT;
  int n = g * 256 + j * 16 + (lane & 15);
  int k = kt * 32 + ((lane >> 4) << 4) + e;
  float v = (k < K && n < N) ? W[(size_t)k * N + n] : 0.f;
  Bp[i] = f2bf(v);
}

// ---- f32 [M,Kin] -> bf16 [M,Kp] (zero pad cols) ----
__global__ void conv_bf16_kernel(const float* __restrict__ in, uint16_t* __restrict__ out,
                                 int Kin, int Kp, int total) {
  int i = blockIdx.x * 256 + threadIdx.x;
  if (i >= total) return;
  int row = i / Kp, col = i % Kp;
  out[i] = f2bf(col < Kin ? in[(size_t)row * Kin + col] : 0.f);
}

// ---- LayerNorm over D=1024 with fused activation; in-place safe ----
__global__ void ln_kernel(const float* __restrict__ in, const float* __restrict__ g,
                          const float* __restrict__ be, float* __restrict__ out, int act) {
  __shared__ float s1[256], s2[256];
  int row = blockIdx.x, tid = threadIdx.x;
  const float* p = in + (size_t)row * DM + tid * 4;
  float x0 = p[0], x1 = p[1], x2 = p[2], x3 = p[3];
  s1[tid] = x0 + x1 + x2 + x3;
  s2[tid] = x0 * x0 + x1 * x1 + x2 * x2 + x3 * x3;
  __syncthreads();
  for (int o = 128; o > 0; o >>= 1) {
    if (tid < o) { s1[tid] += s1[tid + o]; s2[tid] += s2[tid + o]; }
    __syncthreads();
  }
  float mean = s1[0] * (1.f / DM);
  float var  = s2[0] * (1.f / DM) - mean * mean;
  float rs   = rsqrtf(var + 1e-5f);
  float xs[4] = {x0, x1, x2, x3};
  float* q = out + (size_t)row * DM + tid * 4;
  int c0 = tid * 4;
#pragma unroll
  for (int j = 0; j < 4; ++j) {
    float y = (xs[j] - mean) * rs * g[c0 + j] + be[c0 + j];
    if (act == 1)      y = 1.f / (1.f + __expf(-y));
    else if (act == 2) y = tanhf(y);
    q[j] = y;
  }
}

// ---- kv = k*v (b *= a) ----
__global__ void mul_kernel(const float* __restrict__ a, float* __restrict__ b, int total) {
  int i = blockIdx.x * 256 + threadIdx.x;
  if (i < total) b[i] *= a[i];
}

// ---- fused decay-sigmoid + s-scan + s2; in-place over kv buffer ----
__global__ void scan_kernel(const float* __restrict__ dpre, float* __restrict__ kvs2) {
  int ch = blockIdx.x * 256 + threadIdx.x;     // 0 .. BSZ*DM-1
  int b = ch >> 10, d = ch & (DM - 1);
  size_t base = (size_t)b * TLEN * DM + d;
  float s = 0.f;
  for (int t = 0; t < TLEN; ++t) {
    size_t idx = base + (size_t)t * DM;
    float dd  = 1.f / (1.f + __expf(-dpre[idx]));
    float kvv = kvs2[idx];
    s = dd * s + kvv;                  // scan state
    kvs2[idx] = dd * s + kvv;          // re-applied update (faithful to ref)
  }
}

// ---- t = bf16(u + s2) ----
__global__ void addt_kernel(const uint16_t* __restrict__ ubf, const float* __restrict__ s2,
                            uint16_t* __restrict__ tbf, int total) {
  int i = blockIdx.x * 256 + threadIdx.x;
  if (i < total) tbf[i] = f2bf(bf2f(ubf[i]) + s2[i]);
}

// ---- gated h recurrence: h_t = (1-z)*c + z*h_{t-1} ----
__global__ void hrec_kernel(const float* __restrict__ z, const float* __restrict__ c,
                            float* __restrict__ hout) {
  int ch = blockIdx.x * 256 + threadIdx.x;
  int b = ch >> 10, d = ch & (DM - 1);
  size_t base = (size_t)b * TLEN * DM + d;
  float hh = 0.f;
  for (int t = 0; t < TLEN; ++t) {
    size_t idx = base + (size_t)t * DM;
    float zt = z[idx], ct = c[idx];
    hh = (1.f - zt) * ct + zt * hh;
    hout[idx] = hh;
  }
}

// ============================================================
// Host orchestration
// ============================================================
static inline void launch_gemm(const uint16_t* Abf, const uint16_t* Wp, const float* bias,
                               float* Cout, int Kp, int N, int Npad, hipStream_t s) {
  int nG = Npad / 256;
  int blocks = (MROWS / 64) * nG;
  gemm_bf16_kernel<<<blocks, 256, 0, s>>>(Abf, Wp, bias, Cout, Kp, N, nG);
}
static inline void launch_pack(const float* W, uint16_t* Bp, int K, int N, int Kp, int Np,
                               hipStream_t s) {
  int total = Kp * Np;
  pack_w_kernel<<<(total + 255) / 256, 256, 0, s>>>(W, Bp, K, N, Kp, Np);
}

extern "C" void kernel_launch(void* const* d_in, const int* in_sizes, int n_in,
                              void* d_out, int out_size, void* d_ws, size_t ws_size,
                              hipStream_t stream) {
  (void)in_sizes; (void)n_in; (void)out_size; (void)ws_size;
  const float* x = (const float*)d_in[0];
  auto P = [&](int l, int j) -> const float* { return (const float*)d_in[1 + l * 18 + j]; };
  // per-layer param order: Win(0) bin(1) Wz(2) bz(3) Wk(4) bk(5) Wv(6) bv(7)
  //                        Wh(8) bh(9) Wd(10) bd(11) g_in(12) be_in(13)
  //                        g_z(14) be_z(15) g_h(16) be_h(17)
  const float* Wout = (const float*)d_in[1 + 2 * 18 + 0];
  const float* bout = (const float*)d_in[1 + 2 * 18 + 1];
  float* out = (float*)d_out;

  // ---- scratch layout ----
  char* ws = (char*)d_ws;
  size_t off = 0;
  auto carve = [&](size_t bytes) -> void* {
    void* p = ws + off;
    off += (bytes + 255) & ~(size_t)255;
    return p;
  };
  float*    U     = (float*)   carve((size_t)MROWS * DM * 4);      // u (f32)
  float*    A     = (float*)   carve((size_t)MROWS * DM * 4);
  float*    Bf    = (float*)   carve((size_t)MROWS * DM * 4);
  float*    Cf    = (float*)   carve((size_t)MROWS * DM * 4);      // layer output h
  uint16_t* ubf   = (uint16_t*)carve((size_t)MROWS * DM * 2);      // u in bf16
  uint16_t* actbf = (uint16_t*)carve((size_t)MROWS * DM * 2);      // x / t in bf16
  uint16_t* wpack = (uint16_t*)carve((size_t)DM * 5120 * 2);       // packed weights

  const int totMD = MROWS * DM;
  const int eblk  = (totMD + 255) / 256;

  const float* xin = x;
  int din = INDIM;
  for (int l = 0; l < 2; ++l) {
    int Kp0 = (din + 31) & ~31;                   // 96 for layer 0, 1024 after
    int totA = MROWS * Kp0;
    conv_bf16_kernel<<<(totA + 255) / 256, 256, 0, stream>>>(xin, actbf, din, Kp0, totA);

    // u = LN(x @ Win + bin)
    launch_pack(P(l, 0), wpack, din, DM, Kp0, DM, stream);
    launch_gemm(actbf, wpack, P(l, 1), A, Kp0, DM, DM, stream);
    ln_kernel<<<MROWS, 256, 0, stream>>>(A, P(l, 12), P(l, 13), U, 0);
    conv_bf16_kernel<<<eblk, 256, 0, stream>>>(U, ubf, DM, DM, totMD);

    // k, v, kv
    launch_pack(P(l, 4), wpack, DM, DM, DM, DM, stream);
    launch_gemm(ubf, wpack, P(l, 5), A, DM, DM, DM, stream);          // k -> A
    launch_pack(P(l, 6), wpack, DM, DM, DM, DM, stream);
    launch_gemm(ubf, wpack, P(l, 7), Bf, DM, DM, DM, stream);         // v -> B
    mul_kernel<<<eblk, 256, 0, stream>>>(A, Bf, totMD);               // B = k*v

    // decay pre-activation, fused sigmoid + scan + s2 (B becomes s2)
    launch_pack(P(l, 10), wpack, DM, DM, DM, DM, stream);
    launch_gemm(ubf, wpack, P(l, 11), A, DM, DM, DM, stream);         // dpre -> A
    scan_kernel<<<(BSZ * DM) / 256, 256, 0, stream>>>(A, Bf);

    // t = bf16(u + s2)
    addt_kernel<<<eblk, 256, 0, stream>>>(ubf, Bf, actbf, totMD);

    // z = sigmoid(LN(u @ Wz + bz))  (in-place LN in B)
    launch_pack(P(l, 2), wpack, DM, DM, DM, DM, stream);
    launch_gemm(ubf, wpack, P(l, 3), Bf, DM, DM, DM, stream);
    ln_kernel<<<MROWS, 256, 0, stream>>>(Bf, P(l, 14), P(l, 15), Bf, 1);

    // c = tanh(LN(t @ Wh + bh))  (in-place LN in A)
    launch_pack(P(l, 8), wpack, DM, DM, DM, DM, stream);
    launch_gemm(actbf, wpack, P(l, 9), A, DM, DM, DM, stream);
    ln_kernel<<<MROWS, 256, 0, stream>>>(A, P(l, 16), P(l, 17), A, 2);

    // h recurrence -> C (next layer input)
    hrec_kernel<<<(BSZ * DM) / 256, 256, 0, stream>>>(Bf, A, Cf);

    xin = Cf;
    din = DM;
  }

  // logits = h @ Wout + bout  (V=5000 padded to 5120 = 20 N-groups of 256)
  conv_bf16_kernel<<<eblk, 256, 0, stream>>>(Cf, ubf, DM, DM, totMD);
  const int Vpad = 5120;
  launch_pack(Wout, wpack, DM, VOUT, DM, Vpad, stream);
  launch_gemm(ubf, wpack, bout, out, DM, VOUT, Vpad, stream);
}
